// Attention_88270167868034
// MI455X (gfx1250) — matrix operand order repaired and checked by reading, not compiled, not run on hardware
//
#include <hip/hip_runtime.h>
#include <hip/hip_bf16.h>

#define N_TOK 8192
#define DM    256

typedef __attribute__((ext_vector_type(16))) __bf16 v16bf;
typedef __attribute__((ext_vector_type(8)))  __bf16 v8bf;
typedef __attribute__((ext_vector_type(8)))  float  v8f;

// A-layout (16x32 bf16, M per lane, K packed in VGPRs):
// lane l holds row M=l%16; K[16h .. 16h+15] contiguous  (h = l/16)
// -> two contiguous 16B loads at +0 and +8 elements.
__device__ __forceinline__ v16bf ldA(const __bf16* p0) {
    union { v16bf v; v8bf h[2]; } u;
    u.h[0] = *(const v8bf*)(p0);
    u.h[1] = *(const v8bf*)(p0 + 8);
    return u.v;
}
// B-layout (32x16 bf16, N per lane, K packed in VGPRs):
// lane l holds col N=l%16, K[16h..16h+15] contiguous -> 32B contiguous.
__device__ __forceinline__ v16bf ldB(const __bf16* p0) {
    union { v16bf v; v8bf h[2]; } u;
    u.h[0] = *(const v8bf*)(p0);
    u.h[1] = *(const v8bf*)(p0 + 8);
    return u.v;
}

// wave32 ballot of a predicate (bit per lane), with portable fallback
__device__ __forceinline__ unsigned ballot32(bool pred, int lane) {
#if __has_builtin(__builtin_amdgcn_ballot_w32)
    return __builtin_amdgcn_ballot_w32(pred);
#else
    unsigned bit = pred ? (1u << lane) : 0u;
    #pragma unroll
    for (int off = 1; off < 32; off <<= 1)
        bit |= (unsigned)__shfl_xor((int)bit, off, 32);
    return bit;
#endif
}

__global__ __launch_bounds__(256) void cvt_x_kernel(const float* __restrict__ x,
                                                    __bf16* __restrict__ xb) {
    size_t i = (size_t)blockIdx.x * 256 + threadIdx.x;
    xb[i] = (__bf16)x[i];
}

// value = x @ Vw^T + Vb, stored TRANSPOSED in bf16: vbT[d][n]
__global__ __launch_bounds__(32) void value_kernel(const __bf16* __restrict__ xb,
                                                   const float* __restrict__ Vw,
                                                   const float* __restrict__ Vb,
                                                   __bf16* __restrict__ vbT) {
    const int lane = threadIdx.x;
    const int rr = lane & 15, h = lane >> 4;
    const int d0 = blockIdx.x * 16;   // output feature tile
    const int m0 = blockIdx.y * 16;   // token tile
    v8f acc = {};
    #pragma unroll
    for (int kc = 0; kc < 8; ++kc) {
        v16bf a = ldA(xb + (size_t)(m0 + rr) * DM + kc * 32 + 16 * h);
        const float* wp = Vw + (size_t)(d0 + rr) * DM + kc * 32 + 16 * h;
        v16bf b;
        #pragma unroll
        for (int e = 0; e < 16; ++e) b[e] = (__bf16)wp[e];
        acc = __builtin_amdgcn_wmma_f32_16x16x32_bf16(false, a, false, b,
                                                      (short)0, acc, false, false);
    }
    float bias = Vb[d0 + rr];
    v8bf o;
    #pragma unroll
    for (int r = 0; r < 8; ++r) o[r] = (__bf16)(acc[r] + bias);
    // D element (M=r+8h, N=rr): vbT row d = d0+rr, cols m0+8h .. m0+8h+7 contiguous
    *(v8bf*)(vbT + (size_t)(d0 + rr) * N_TOK + m0 + 8 * h) = o;
}

// One block = 16 query rows. 8 waves, each owns a 1024-column key slice.
// Sweep1: scores (WMMA bf16) + mask -> online softmax stats, adj ballot-packed to LDS bitmask.
// Sweep2: recompute scores, p = exp(s-m)/sum -> write p_attn, and accumulate
//         Vs = p @ value via a wave-private LDS p-tile round-trip (ordered with
//         s_wait_dscnt, no block barrier) + WMMA against transposed value.
__global__ __launch_bounds__(256) void attn_kernel(const __bf16* __restrict__ xb,
                                                   const int* __restrict__ adj,
                                                   const __bf16* __restrict__ vbT,
                                                   float* __restrict__ outV,
                                                   float* __restrict__ outP) {
    __shared__ __bf16   lds_xa[16 * 256];     // 8KB  query rows (bf16)
    __shared__ unsigned lds_mask[512 * 8];    // 16KB adj bitmask for this row block
    __shared__ __bf16   lds_pt[8][16 * 32];   // 8KB  per-wave p tiles (16x32)
    __shared__ float    lds_wm[8][16];        // per-wave row max
    __shared__ float    lds_wsum[8][16];      // per-wave row sumexp
    __shared__ float    lds_fm[16];           // final row max
    __shared__ float    lds_fi[16];           // final 1/sum
    __shared__ float    lds_vs[16 * 256];     // 16KB Vs accumulator

    const int tid  = threadIdx.x;
    const int lane = tid & 31;
    const int wv   = tid >> 5;
    const int rr   = lane & 15;
    const int h    = lane >> 4;
    const int r0   = blockIdx.x * 16;
    const int cb   = wv * 1024;               // this wave's key-column slice
    const float scale = 0.0625f;              // 1/sqrt(256)

    #pragma unroll
    for (int i = 0; i < 16; ++i) lds_vs[tid + 256 * i] = 0.0f;

    // Stage this row-block of xb into LDS with CDNA5 async global->LDS copies.
    // Each thread moves 32B: dsaddr = VDST_vgpr + inst_offset, gaddr = SADDR + VADDR + inst_offset.
    {
        const __bf16* gsrc = xb + (size_t)r0 * DM;
        unsigned voff  = (unsigned)tid * 32u;
        unsigned ldsa  = (unsigned)(size_t)(&lds_xa[0]) + voff;
        asm volatile("global_load_async_to_lds_b128 %0, %1, %2"
                     :: "v"(ldsa), "v"(voff), "s"(gsrc) : "memory");
        asm volatile("global_load_async_to_lds_b128 %0, %1, %2 offset:16"
                     :: "v"(ldsa), "v"(voff), "s"(gsrc) : "memory");
        asm volatile("s_wait_asynccnt 0x0" ::: "memory");
    }
    __syncthreads();

    float m[8], s[8];
    #pragma unroll
    for (int r = 0; r < 8; ++r) { m[r] = -1e30f; s[r] = 0.0f; }

    // ---------------- sweep 1: stats + mask packing ----------------
    for (int kt = 0; kt < 64; ++kt) {
        // prefetch the adj tile 8 iterations ahead (stays inside this row block)
        if (kt < 56)
            __builtin_prefetch(adj + (size_t)(r0 + rr) * N_TOK + cb + (kt + 8) * 16 + 8 * h, 0, 0);
        v8f acc = {};
        const __bf16* brow = xb + (size_t)(cb + kt * 16 + rr) * DM;
        const __bf16* arow = lds_xa + rr * DM;
        #pragma unroll
        for (int kc = 0; kc < 8; ++kc) {
            v16bf a = ldA(arow + kc * 32 + 16 * h);
            v16bf b = ldB(brow + kc * 32 + 16 * h);
            acc = __builtin_amdgcn_wmma_f32_16x16x32_bf16(false, a, false, b,
                                                          (short)0, acc, false, false);
        }
        #pragma unroll
        for (int r = 0; r < 8; ++r) {
            int av = adj[(size_t)(r0 + r + 8 * h) * N_TOK + cb + kt * 16 + rr];
            unsigned word = ballot32(av != 0, lane);
            if (lane == 0) lds_mask[(wv * 64 + kt) * 8 + r] = word;
            float v = (av != 0) ? acc[r] * scale : -1e9f;
            float nm = fmaxf(m[r], v);
            s[r] = s[r] * __expf(m[r] - nm) + __expf(v - nm);
            m[r] = nm;
        }
    }
    // reduce stats over the 16 lanes holding the same row
    #pragma unroll
    for (int r = 0; r < 8; ++r) {
        #pragma unroll
        for (int off = 1; off < 16; off <<= 1) {
            float mo = __shfl_xor(m[r], off, 32);
            float so = __shfl_xor(s[r], off, 32);
            float nm = fmaxf(m[r], mo);
            s[r] = s[r] * __expf(m[r] - nm) + so * __expf(mo - nm);
            m[r] = nm;
        }
    }
    if (rr == 0) {
        #pragma unroll
        for (int r = 0; r < 8; ++r) {
            lds_wm[wv][r + 8 * h] = m[r];
            lds_wsum[wv][r + 8 * h] = s[r];
        }
    }
    __syncthreads();
    if (tid < 16) {   // merge the 8 wave-partials per row
        float fm = -1e30f, fs = 0.0f;
        #pragma unroll
        for (int w = 0; w < 8; ++w) {
            float mo = lds_wm[w][tid], so = lds_wsum[w][tid];
            float nm = fmaxf(fm, mo);
            fs = fs * __expf(fm - nm) + so * __expf(mo - nm);
            fm = nm;
        }
        lds_fm[tid] = fm;
        lds_fi[tid] = 1.0f / fs;
    }
    __syncthreads();

    float fm8[8], fi8[8];
    #pragma unroll
    for (int r = 0; r < 8; ++r) { fm8[r] = lds_fm[r + 8 * h]; fi8[r] = lds_fi[r + 8 * h]; }

    v8f accv[16] = {};   // Vs accumulator: 16 feature tiles x 8 rows

    // ---------------- sweep 2: p_attn + Vs (no block barriers: p-tile is wave-private,
    //                  DS ops are in-order per wave; s_wait_dscnt orders store->load) ----
    for (int ktp = 0; ktp < 32; ++ktp) {
        #pragma unroll
        for (int q = 0; q < 2; ++q) {
            const int kt = ktp * 2 + q;
            v8f acc = {};
            const __bf16* brow = xb + (size_t)(cb + kt * 16 + rr) * DM;
            const __bf16* arow = lds_xa + rr * DM;
            #pragma unroll
            for (int kc = 0; kc < 8; ++kc) {
                v16bf a = ldA(arow + kc * 32 + 16 * h);
                v16bf b = ldB(brow + kc * 32 + 16 * h);
                acc = __builtin_amdgcn_wmma_f32_16x16x32_bf16(false, a, false, b,
                                                              (short)0, acc, false, false);
            }
            #pragma unroll
            for (int r = 0; r < 8; ++r) {
                unsigned word = lds_mask[(wv * 64 + kt) * 8 + r];
                float v = ((word >> lane) & 1u) ? acc[r] * scale : -1e9f;
                float p = __expf(v - fm8[r]) * fi8[r];
                outP[(size_t)(r0 + r + 8 * h) * N_TOK + cb + kt * 16 + rr] = p;
                lds_pt[wv][(r + 8 * h) * 32 + q * 16 + rr] = (__bf16)p;
            }
        }
        // wave-local ordering: all p-tile DS stores complete before the A-layout re-read
        asm volatile("s_wait_dscnt 0x0" ::: "memory");
        v16bf ap = ldA(&lds_pt[wv][rr * 32 + 16 * h]);
        #pragma unroll
        for (int dt = 0; dt < 16; ++dt) {
            const __bf16* vrow = vbT + (size_t)(dt * 16 + rr) * N_TOK + cb + ktp * 32 + 16 * h;
            v16bf bv = ldB(vrow);
            accv[dt] = __builtin_amdgcn_wmma_f32_16x16x32_bf16(false, ap, false, bv,
                                                               (short)0, accv[dt], false, false);
        }
        // compiler fence: next iteration's p-tile stores must not be hoisted above the reads
        asm volatile("" ::: "memory");
    }

    // reduce partial Vs across the 8 waves
    #pragma unroll
    for (int dt = 0; dt < 16; ++dt)
        #pragma unroll
        for (int r = 0; r < 8; ++r)
            atomicAdd(&lds_vs[(r + 8 * h) * DM + dt * 16 + rr], accv[dt][r]);
    __syncthreads();
    #pragma unroll
    for (int i = 0; i < 16; ++i) {
        int idx = tid + 256 * i;
        int row = idx >> 8, col = idx & 255;
        outV[(size_t)(r0 + row) * DM + col] = lds_vs[idx];
    }
}

extern "C" void kernel_launch(void* const* d_in, const int* in_sizes, int n_in,
                              void* d_out, int out_size, void* d_ws, size_t ws_size,
                              hipStream_t stream) {
    (void)in_sizes; (void)n_in; (void)out_size; (void)ws_size;
    const float* x   = (const float*)d_in[0];
    const int*   adj = (const int*)d_in[1];
    const float* Vw  = (const float*)d_in[2];
    const float* Vb  = (const float*)d_in[3];
    float* outV = (float*)d_out;                      // Vs [8192,256]
    float* outP = outV + (size_t)N_TOK * DM;          // p_attn [8192,8192]
    __bf16* xb  = (__bf16*)d_ws;                      // 4MB  x in bf16
    __bf16* vbT = xb + (size_t)N_TOK * DM;            // 4MB  value^T in bf16

    cvt_x_kernel<<<(N_TOK * DM) / 256, 256, 0, stream>>>(x, xb);
    value_kernel<<<dim3(DM / 16, N_TOK / 16), 32, 0, stream>>>(xb, Vw, Vb, vbT);
    attn_kernel<<<N_TOK / 16, 256, 0, stream>>>(xb, adj, vbT, outV, outP);
}
